// MultiheadLinearAttention_33758442946703
// MI455X (gfx1250) — compile-verified, matrix operand order checked by reading, not run on hardware
//
#include <hip/hip_runtime.h>
#include <hip/hip_bf16.h>

#define BS_   32768
#define E_    1024
#define H_    16
#define D_    64
#define P_    32

typedef __attribute__((ext_vector_type(16))) __bf16 v16bf;
typedef __attribute__((ext_vector_type(8)))  __bf16 v8bf;
typedef __attribute__((ext_vector_type(8)))  float  v8f;

// ---------------------------------------------------------------------------
// Tiled bf16 WMMA GEMM:  C[M,N] = A[M,K] * B[K,N] + bias[N]
//   A: f32 (converted to bf16 while staging) or bf16 (template AT)
//   B: always f32 weights, converted while staging (stored transposed in LDS)
//   Workgroup tile 128x128, BK=32, 8 waves (4M x 2N), wave tile 32x64.
//   Double-buffered LDS (ping-pong): global loads for K-step i+1 overlap the
//   8 WMMAs of K-step i; one barrier per iteration.
// ---------------------------------------------------------------------------
template <typename AT, bool OUT_BF16>
__global__ __launch_bounds__(256)
void gemm_wmma_bf16(const AT* __restrict__ A, const float* __restrict__ B,
                    const float* __restrict__ bias, void* __restrict__ Cout,
                    int M, int N, int K) {
    constexpr int BM = 128, BN = 128, BK = 32;
    __shared__ __bf16 As[2][BM][BK];    // 2 x 8 KB
    __shared__ __bf16 BsT[2][BN][BK];   // 2 x 8 KB, B transposed: BsT[n][k]

    const int tid  = threadIdx.x;
    const int lane = tid & 31;
    const int w    = tid >> 5;          // wave 0..7
    const int wm   = w & 3;             // wave M index (0..3) -> 32 rows each
    const int wn   = w >> 2;            // wave N index (0..1) -> 64 cols each
    const int sub  = lane & 15;
    const int kb   = (lane >> 4) * 8;   // K sub-base per half-wave
    const int mBase = blockIdx.y * BM;
    const int nBase = blockIdx.x * BN;

    v8f acc[2][4] = {};

    // global->register staging
    float4 aRegF[4];
    v8bf   aRegH[2];
    float4 bReg[4];

    auto g_load = [&](int kt) {
        if constexpr (sizeof(AT) == 4) {          // f32 A
            #pragma unroll
            for (int i = 0; i < 4; ++i) {
                int vi = tid + i * 256;           // 1024 float4 (128x32)
                int r  = vi >> 3;
                int c4 = (vi & 7) * 4;
                aRegF[i] = *reinterpret_cast<const float4*>(
                    &A[(size_t)(mBase + r) * K + kt + c4]);
            }
        } else {                                  // bf16 A
            #pragma unroll
            for (int i = 0; i < 2; ++i) {
                int vi = tid + i * 256;           // 512 x 8-bf16 (128x32)
                int r  = vi >> 2;
                int c8 = (vi & 3) * 8;
                aRegH[i] = *reinterpret_cast<const v8bf*>(
                    &A[(size_t)(mBase + r) * K + kt + c8]);
            }
        }
        #pragma unroll
        for (int i = 0; i < 4; ++i) {
            int vi = tid + i * 256;               // 1024 float4 (32x128)
            int r  = vi >> 5;
            int c4 = (vi & 31) * 4;
            bReg[i] = *reinterpret_cast<const float4*>(
                &B[(size_t)(kt + r) * N + nBase + c4]);
        }
    };

    auto lds_store = [&](int buf) {
        if constexpr (sizeof(AT) == 4) {
            #pragma unroll
            for (int i = 0; i < 4; ++i) {
                int vi = tid + i * 256;
                int r  = vi >> 3;
                int c4 = (vi & 7) * 4;
                As[buf][r][c4 + 0] = (__bf16)aRegF[i].x;
                As[buf][r][c4 + 1] = (__bf16)aRegF[i].y;
                As[buf][r][c4 + 2] = (__bf16)aRegF[i].z;
                As[buf][r][c4 + 3] = (__bf16)aRegF[i].w;
            }
        } else {
            #pragma unroll
            for (int i = 0; i < 2; ++i) {
                int vi = tid + i * 256;
                int r  = vi >> 2;
                int c8 = (vi & 3) * 8;
                *reinterpret_cast<v8bf*>(&As[buf][r][c8]) = aRegH[i];
            }
        }
        #pragma unroll
        for (int i = 0; i < 4; ++i) {
            int vi = tid + i * 256;
            int r  = vi >> 5;
            int c4 = (vi & 31) * 4;
            BsT[buf][c4 + 0][r] = (__bf16)bReg[i].x;
            BsT[buf][c4 + 1][r] = (__bf16)bReg[i].y;
            BsT[buf][c4 + 2][r] = (__bf16)bReg[i].z;
            BsT[buf][c4 + 3][r] = (__bf16)bReg[i].w;
        }
    };

    // prologue: stage K-tile 0 into buffer 0
    g_load(0);
    lds_store(0);
    __syncthreads();

    const int nK = K / BK;
    for (int ki = 0; ki < nK; ++ki) {
        const int buf = ki & 1;
        // overlap: fetch next K-tile from global while WMMAs run on this one
        if (ki + 1 < nK) g_load((ki + 1) * BK);

        // fragments (CDNA5 16-bit A/B layouts; contiguous -> ds_load_b128)
        v16bf fa[2], fb[4];
        #pragma unroll
        for (int mf = 0; mf < 2; ++mf) {
            int row = wm * 32 + mf * 16 + sub;
            #pragma unroll
            for (int i = 0; i < 8; ++i) {
                fa[mf][i]     = As[buf][row][kb + i];
                fa[mf][8 + i] = As[buf][row][16 + kb + i];
            }
        }
        #pragma unroll
        for (int nf = 0; nf < 4; ++nf) {
            int col = wn * 64 + nf * 16 + sub;
            #pragma unroll
            for (int i = 0; i < 8; ++i) {
                fb[nf][i]     = BsT[buf][col][kb + i];
                fb[nf][8 + i] = BsT[buf][col][16 + kb + i];
            }
        }
        #pragma unroll
        for (int mf = 0; mf < 2; ++mf)
            #pragma unroll
            for (int nf = 0; nf < 4; ++nf)
                acc[mf][nf] = __builtin_amdgcn_wmma_f32_16x16x32_bf16(
                    false, fa[mf], false, fb[nf],
                    (short)0, acc[mf][nf], false, false);

        if (ki + 1 < nK) lds_store(buf ^ 1);
        __syncthreads();
    }

    // epilogue: bias add + store (C/D layout: lane half -> M 0..7 / 8..15)
    const int half = lane >> 4;
    #pragma unroll
    for (int mf = 0; mf < 2; ++mf)
        #pragma unroll
        for (int nf = 0; nf < 4; ++nf) {
            int col = nBase + wn * 64 + nf * 16 + sub;
            float bcol = bias[col];
            #pragma unroll
            for (int r = 0; r < 8; ++r) {
                int row = mBase + wm * 32 + mf * 16 + half * 8 + r;
                float val = acc[mf][nf][r] + bcol;
                if constexpr (OUT_BF16)
                    ((__bf16*)Cout)[(size_t)row * N + col] = (__bf16)val;
                else
                    ((float*)Cout)[(size_t)row * N + col] = val;
            }
        }
}

// ---------------------------------------------------------------------------
// Fused middle stage: one wave per batch row (P=32 == wave32 lanes).
//   qk = <q,k> per head, bias = attn_bias @ Wfe + bfe, softmax over P,
//   af = <attn,f>, g = af * v  (bf16, written into the q buffer).
// ---------------------------------------------------------------------------
__global__ __launch_bounds__(256)
void fused_attn_mid(const __bf16* __restrict__ q, const __bf16* __restrict__ k,
                    const __bf16* __restrict__ v, const __bf16* __restrict__ e,
                    const __bf16* __restrict__ f,
                    const float* __restrict__ attn_bias,
                    const float* __restrict__ Wfe, const float* __restrict__ bfe,
                    __bf16* __restrict__ g) {
    __shared__ float sWfe[H_ * H_ * P_];   // 16 x 512 = 32 KB
    __shared__ float sbfe[H_ * P_];        // 2 KB
    const int tid = threadIdx.x;
    for (int i = tid; i < H_ * H_ * P_; i += 256) sWfe[i] = Wfe[i];
    for (int i = tid; i < H_ * P_; i += 256)      sbfe[i] = bfe[i];
    __syncthreads();

    const int w = tid >> 5;
    const int l = tid & 31;
    const int b = blockIdx.x * 8 + w;
    const float scale = 0.125f;            // 1/sqrt(64)

    float ab[H_];
    #pragma unroll
    for (int j = 0; j < H_; ++j) ab[j] = attn_bias[(size_t)b * H_ + j];

    const size_t bqv = (size_t)b * (H_ * D_);   // *1024
    const size_t bef = (size_t)b * (H_ * P_);   // *512

    for (int h = 0; h < H_; ++h) {
        const size_t ho = bqv + h * D_;
        // qk dot over D=64 (2 elements per lane), wave-wide reduce
        float qk = (float)q[ho + l] * (float)k[ho + l]
                 + (float)q[ho + 32 + l] * (float)k[ho + 32 + l];
        #pragma unroll
        for (int off = 16; off; off >>= 1) qk += __shfl_xor(qk, off, 32);

        const int ep = h * P_ + l;
        float bias_l = sbfe[ep];
        #pragma unroll
        for (int j = 0; j < H_; ++j) bias_l += ab[j] * sWfe[j * (H_ * P_) + ep];

        float logit = scale * qk * (float)e[bef + ep] + bias_l;

        float m = logit;
        #pragma unroll
        for (int off = 16; off; off >>= 1) m = fmaxf(m, __shfl_xor(m, off, 32));
        float ex = __expf(logit - m);
        float s = ex;
        #pragma unroll
        for (int off = 16; off; off >>= 1) s += __shfl_xor(s, off, 32);
        float attn = ex / s;

        float af = attn * (float)f[bef + ep];
        #pragma unroll
        for (int off = 16; off; off >>= 1) af += __shfl_xor(af, off, 32);

        g[ho + l]      = (__bf16)(af * (float)v[ho + l]);
        g[ho + 32 + l] = (__bf16)(af * (float)v[ho + 32 + l]);
    }
}

// ---------------------------------------------------------------------------
extern "C" void kernel_launch(void* const* d_in, const int* in_sizes, int n_in,
                              void* d_out, int out_size, void* d_ws, size_t ws_size,
                              hipStream_t stream) {
    const float* query = (const float*)d_in[0];
    const float* key   = (const float*)d_in[1];
    const float* value = (const float*)d_in[2];
    const float* abias = (const float*)d_in[3];
    const float* Wq = (const float*)d_in[4];  const float* bq  = (const float*)d_in[5];
    const float* Wk = (const float*)d_in[6];  const float* bk  = (const float*)d_in[7];
    const float* Wv = (const float*)d_in[8];  const float* bv  = (const float*)d_in[9];
    const float* We = (const float*)d_in[10]; const float* be  = (const float*)d_in[11];
    const float* Wf = (const float*)d_in[12]; const float* bfv = (const float*)d_in[13];
    const float* Wfe= (const float*)d_in[14]; const float* bfe = (const float*)d_in[15];
    const float* Wo = (const float*)d_in[16]; const float* bo  = (const float*)d_in[17];
    float* out = (float*)d_out;

    // Workspace layout (bf16 intermediates, 256 MiB total; g aliases q).
    __bf16* qb = (__bf16*)d_ws;
    __bf16* kb = qb + (size_t)BS_ * E_;
    __bf16* vb = kb + (size_t)BS_ * E_;
    __bf16* eb = vb + (size_t)BS_ * E_;
    __bf16* fb = eb + (size_t)BS_ * (H_ * P_);

    dim3 blk(256);
    dim3 gE(E_ / 128, BS_ / 128);          // N=1024 GEMMs
    dim3 gP((H_ * P_) / 128, BS_ / 128);   // N=512  GEMMs

    gemm_wmma_bf16<float, true ><<<gE, blk, 0, stream>>>(query, Wq, bq, qb, BS_, E_, E_);
    gemm_wmma_bf16<float, true ><<<gE, blk, 0, stream>>>(key,   Wk, bk, kb, BS_, E_, E_);
    gemm_wmma_bf16<float, true ><<<gE, blk, 0, stream>>>(value, Wv, bv, vb, BS_, E_, E_);
    gemm_wmma_bf16<float, true ><<<gP, blk, 0, stream>>>(key,   We, be, eb, BS_, H_ * P_, E_);
    gemm_wmma_bf16<float, true ><<<gP, blk, 0, stream>>>(value, Wf, bfv, fb, BS_, H_ * P_, E_);

    fused_attn_mid<<<dim3(BS_ / 8), blk, 0, stream>>>(qb, kb, vb, eb, fb,
                                                      abias, Wfe, bfe, /*g=*/qb);

    gemm_wmma_bf16<__bf16, false><<<gE, blk, 0, stream>>>(qb, Wo, bo, out, BS_, E_, E_);
}
